// GQAHead_11115375362970
// MI455X (gfx1250) — matrix operand order repaired and checked by reading, not compiled, not run on hardware
//
#include <hip/hip_runtime.h>
#include <hip/hip_bf16.h>

// ---------------------------------------------------------------------------
// Problem constants (from the reference)
#define BATCH 8
#define TSEQ  2048
#define CDIM  2048
#define HDIM  128

typedef __attribute__((ext_vector_type(16))) __bf16    v16bf;
typedef __attribute__((ext_vector_type(8)))  float     v8f;
typedef __attribute__((ext_vector_type(4)))  float     v4f;
typedef __attribute__((ext_vector_type(4)))  unsigned  v4u;

// One 16-bit A/B WMMA fragment = 8 VGPRs = two 16-byte halves.
struct Frag { v4u lo, hi; };

static __device__ inline v16bf frag_cast(const Frag& f) {
    return __builtin_bit_cast(v16bf, f);
}

static __device__ inline unsigned pack2bf(float lo, float hi) {
    __bf16 a = (__bf16)lo, b = (__bf16)hi;
    unsigned short ua = __builtin_bit_cast(unsigned short, a);
    unsigned short ub = __builtin_bit_cast(unsigned short, b);
    return (unsigned)ua | ((unsigned)ub << 16);
}

// 16-byte bf16 fragment-half load (4 packed pairs, 16B aligned by construction)
static __device__ inline v4u ld4u(const __bf16* p) {
    return *(const v4u*)p;
}

#define LN10000 9.210340371976184f   // ln(10000)

// ---------------------------------------------------------------------------
// Kernel 0: Wq f32 -> bf16 once (L2-resident operand; kills per-wave cvt work)
__global__ void cvt_wq_kernel(const float* __restrict__ Wq, __bf16* __restrict__ wqb)
{
    int i = blockIdx.x * blockDim.x + threadIdx.x;
    if (i < HDIM * CDIM) wqb[i] = (__bf16)Wq[i];
}

// ---------------------------------------------------------------------------
// Kernel 1: q = x @ Wq^T (bf16 WMMA, f32 acc), fused RoPE, bf16 out.
// One wave computes a 16(t) x 128(h) strip; x streamed non-temporally.
__global__ void qproj_rope_kernel(const float* __restrict__ x,
                                  const __bf16* __restrict__ wqb,
                                  __bf16* __restrict__ qrot)
{
    const int wid = threadIdx.x >> 5;
    const int ln  = threadIdx.x & 31;
    const int rt  = blockIdx.x * 4 + wid;          // 0 .. B*T/16-1
    const int b   = rt / (TSEQ / 16);
    const int t0  = (rt % (TSEQ / 16)) * 16;

    const int lrow = ln & 15;
    const int hiA  = (ln >= 16) ? 8  : 0;          // A-layout K base (16-bit A)
    const int hiB  = (ln >= 16) ? 16 : 0;          // unused here (B loads use hiA, same k order as A)
    const int hiD  = (ln >= 16) ? 8  : 0;          // C/D-layout row base

    const float* xrow = x + (size_t)b * TSEQ * CDIM + (size_t)(t0 + lrow) * CDIM;

    v8f acc[8] = {};

    for (int k = 0; k < CDIM; k += 32) {
        // A fragment: 16 rows of x, K = 32 cols of C; two 8-float NT streams.
        const v4f* pa0 = (const v4f*)(xrow + k + hiA);        // c = k+hiA .. +7
        const v4f* pa1 = (const v4f*)(xrow + k + 16 + hiA);   // c = k+16+hiA .. +7
        v4f a0 = __builtin_nontemporal_load(pa0);
        v4f a1 = __builtin_nontemporal_load(pa0 + 1);
        v4f a2 = __builtin_nontemporal_load(pa1);
        v4f a3 = __builtin_nontemporal_load(pa1 + 1);
        Frag af;
        af.lo = (v4u){ pack2bf(a0.x, a0.y), pack2bf(a0.z, a0.w),
                       pack2bf(a1.x, a1.y), pack2bf(a1.z, a1.w) };
        af.hi = (v4u){ pack2bf(a2.x, a2.y), pack2bf(a2.z, a2.w),
                       pack2bf(a3.x, a3.y), pack2bf(a3.z, a3.w) };
        v16bf a = frag_cast(af);
#pragma unroll
        for (int ht = 0; ht < 8; ++ht) {
            // B fragment: B[k,n] = Wq[ht*16+n, c]; bf16, two b128 loads (L2 hits)
            const __bf16* wp = wqb + (size_t)(ht * 16 + lrow) * CDIM + k + hiA;
            Frag bf;
            bf.lo = ld4u(wp);
            bf.hi = ld4u(wp + 16);
            acc[ht] = __builtin_amdgcn_wmma_f32_16x16x32_bf16(
                false, a, false, frag_cast(bf), (short)0, acc[ht], false, false);
        }
    }

    // RoPE on the 16x128 strip. D layout: VGPR d -> row t0+d+hiD, lane -> h col.
#pragma unroll
    for (int ht = 0; ht < 8; ++ht) {
        int h = ht * 16 + lrow;
        int p = h >> 1;
        float inv = __expf(-((float)(2 * p) / 128.0f) * LN10000);
#pragma unroll
        for (int d = 0; d < 8; ++d) {
            int t = t0 + d + hiD;
            float ang = (float)t * inv, sn, cs;
            __sincosf(ang, &sn, &cs);
            float val   = acc[ht][d];
            float other = __shfl_xor(val, 1, 32);   // pair partner (h parity == lane parity)
            float out   = (h & 1) ? (other * sn + val * cs)
                                  : (val * cs - other * sn);
            qrot[((size_t)b * TSEQ + t) * HDIM + h] = (__bf16)out;
        }
    }
}

// ---------------------------------------------------------------------------
// Kernel 2: RoPE on key, bf16 out. One thread per (b,t,pair).
__global__ void rope_k_kernel(const float* __restrict__ key,
                              __bf16* __restrict__ krot)
{
    int i = blockIdx.x * blockDim.x + threadIdx.x;
    const int NP = BATCH * TSEQ * (HDIM / 2);
    if (i >= NP) return;
    int p = i % (HDIM / 2);
    int t = (i / (HDIM / 2)) % TSEQ;
    int b = i / ((HDIM / 2) * TSEQ);

    const float* kp = key + ((size_t)b * TSEQ + t) * HDIM + 2 * p;
    float x0 = kp[0], x1 = kp[1];
    float inv = __expf(-((float)(2 * p) / 128.0f) * LN10000);
    float ang = (float)t * inv, sn, cs;
    __sincosf(ang, &sn, &cs);
    unsigned pkd = pack2bf(x0 * cs - x1 * sn, x0 * sn + x1 * cs);
    *(unsigned*)(krot + ((size_t)b * TSEQ + t) * HDIM + 2 * p) = pkd;
}

// ---------------------------------------------------------------------------
// Kernel 3: transpose value to [B, H, T] bf16 so V^T fragments load contiguously.
__global__ void transpose_v_kernel(const float* __restrict__ val,
                                   __bf16* __restrict__ vT)
{
    int i = blockIdx.x * blockDim.x + threadIdx.x;
    const int N = BATCH * HDIM * TSEQ;
    if (i >= N) return;
    int t = i % TSEQ;
    int h = (i / TSEQ) % HDIM;
    int b = i / (TSEQ * HDIM);
    vT[i] = (__bf16)val[((size_t)b * TSEQ + t) * HDIM + h];
}

// ---------------------------------------------------------------------------
// Kernel 4: flash attention, one wave per 16-query tile, 32-key chunks.
// S^T = K * Q^T (softmax reduction lane-local), O^T = V^T * P^T.
__global__ void attn_kernel(const __bf16* __restrict__ qrot,
                            const __bf16* __restrict__ krot,
                            const __bf16* __restrict__ vT,
                            float* __restrict__ out)
{
    const float scale = 0.08838834764831845f;   // 1/sqrt(128)
    const int wid = threadIdx.x >> 5;
    const int ln  = threadIdx.x & 31;
    const int qt  = blockIdx.x * 4 + wid;       // 0 .. B*T/16-1
    const int b   = qt / (TSEQ / 16);
    const int q0  = (qt % (TSEQ / 16)) * 16;

    const int lq  = ln & 15;
    const int qg  = q0 + lq;                    // this lane's query column
    const int hiA = (ln >= 16) ? 8  : 0;
    const int hiB = (ln >= 16) ? 16 : 0;
    const int hi8 = (ln >= 16) ? 8  : 0;

    const __bf16* qb = qrot + (size_t)b * TSEQ * HDIM;
    const __bf16* kb = krot + (size_t)b * TSEQ * HDIM;
    const __bf16* vb = vT   + (size_t)b * HDIM * TSEQ;

    // Q^T B-fragments: B[k,n] = q[q0+n, 32f+k]; loaded once (4 x 8 VGPRs).
    Frag qf[4];
#pragma unroll
    for (int f = 0; f < 4; ++f) {
        const __bf16* qp = qb + (size_t)qg * HDIM + 32 * f + hiA;
        qf[f].lo = ld4u(qp);
        qf[f].hi = ld4u(qp + 16);
    }

    v8f accO[8] = {};                           // O^T: 8 h-tiles x (16h x 16q)
    float m = -1e30f, l = 0.0f;

    const int nch = (q0 + 16 + 31) / 32;
    for (int c = 0; c < nch; ++c) {
        const int sb = 32 * c;

        // --- S^T for 32 keys: two 16x16 tiles, K-dim = H in 4 steps of 32 ---
        v8f S0 = {}, S1 = {};
#pragma unroll
        for (int f = 0; f < 4; ++f) {
            const __bf16* k0p = kb + (size_t)(sb + lq) * HDIM + 32 * f;
            const __bf16* k1p = k0p + 16 * HDIM;
            Frag ka0, ka1;                      // A: rows = keys, K = h
            ka0.lo = ld4u(k0p + hiA);      ka0.hi = ld4u(k0p + 16 + hiA);
            ka1.lo = ld4u(k1p + hiA);      ka1.hi = ld4u(k1p + 16 + hiA);
            S0 = __builtin_amdgcn_wmma_f32_16x16x32_bf16(
                false, frag_cast(ka0), false, frag_cast(qf[f]), (short)0, S0, false, false);
            S1 = __builtin_amdgcn_wmma_f32_16x16x32_bf16(
                false, frag_cast(ka1), false, frag_cast(qf[f]), (short)0, S1, false, false);
        }

        // --- scale + causal mask (key index = sb + d + hi8 [+16]) ---
        float p0[8], p1[8];
#pragma unroll
        for (int d = 0; d < 8; ++d) {
            int k0 = sb + d + hi8;
            float s0 = S0[d] * scale; if (k0      > qg) s0 = -1e30f;
            float s1 = S1[d] * scale; if (k0 + 16 > qg) s1 = -1e30f;
            p0[d] = s0; p1[d] = s1;
        }

        // --- online softmax: per-column max/sum = 8 regs + shfl_xor(16) ---
        float mt = -1e30f;
#pragma unroll
        for (int d = 0; d < 8; ++d) mt = fmaxf(mt, fmaxf(p0[d], p1[d]));
        mt = fmaxf(mt, __shfl_xor(mt, 16, 32));
        float mnew = fmaxf(m, mt);
        float corr = __expf(m - mnew);
        float rs = 0.0f;
#pragma unroll
        for (int d = 0; d < 8; ++d) {
            p0[d] = __expf(p0[d] - mnew);
            p1[d] = __expf(p1[d] - mnew);
            rs += p0[d] + p1[d];
        }
        rs += __shfl_xor(rs, 16, 32);
        l = l * corr + rs;
        m = mnew;
#pragma unroll
        for (int ht = 0; ht < 8; ++ht) accO[ht] = accO[ht] * corr;

        // --- repack P^T (D layout) into a B fragment [32s x 16q] ---
        // lane<16 needs s=0..15 (own S0 rows 0..7 + partner S0 rows 8..15);
        // lane>=16 needs s=16..31 (partner S1 rows 16..23 + own S1 rows 24..31).
        float lo8[8], hv8[8];
#pragma unroll
        for (int d = 0; d < 8; ++d) {
            float send = (ln >= 16) ? p0[d] : p1[d];
            float recv = __shfl_xor(send, 16, 32);
            lo8[d] = (ln < 16) ? p0[d] : recv;
            hv8[d] = (ln < 16) ? recv  : p1[d];
        }
        Frag pf;
        pf.lo = (v4u){ pack2bf(lo8[0], lo8[1]), pack2bf(lo8[2], lo8[3]),
                       pack2bf(lo8[4], lo8[5]), pack2bf(lo8[6], lo8[7]) };
        pf.hi = (v4u){ pack2bf(hv8[0], hv8[1]), pack2bf(hv8[2], hv8[3]),
                       pack2bf(hv8[4], hv8[5]), pack2bf(hv8[6], hv8[7]) };
        v16bf pb = frag_cast(pf);

        // --- O^T += V^T * P^T : A rows = h, K = s (contiguous in vT) ---
#pragma unroll
        for (int ht = 0; ht < 8; ++ht) {
            const __bf16* vp = vb + (size_t)(ht * 16 + lq) * TSEQ + sb;
            Frag va;
            va.lo = ld4u(vp + hiB);
            va.hi = ld4u(vp + hiB + 8);
            accO[ht] = __builtin_amdgcn_wmma_f32_16x16x32_bf16(
                false, frag_cast(va), false, pb, (short)0, accO[ht], false, false);
        }
    }

    // --- write out[b, q, h] = accO / l (write-once -> non-temporal) ---
    float rl = 1.0f / l;
#pragma unroll
    for (int ht = 0; ht < 8; ++ht)
#pragma unroll
        for (int d = 0; d < 8; ++d) {
            int h = ht * 16 + d + hi8;
            __builtin_nontemporal_store(accO[ht][d] * rl,
                out + ((size_t)b * TSEQ + qg) * HDIM + h);
        }
}

// ---------------------------------------------------------------------------
extern "C" void kernel_launch(void* const* d_in, const int* in_sizes, int n_in,
                              void* d_out, int out_size, void* d_ws, size_t ws_size,
                              hipStream_t stream) {
    const float* x   = (const float*)d_in[0];
    const float* key = (const float*)d_in[1];
    const float* val = (const float*)d_in[2];
    const float* Wq  = (const float*)d_in[3];
    float* out = (float*)d_out;

    const size_t NE = (size_t)BATCH * TSEQ * HDIM;   // 2,097,152 elements
    __bf16* qrot = (__bf16*)d_ws;                    // 4 MB
    __bf16* krot = qrot + NE;                        // 4 MB
    __bf16* vT   = krot + NE;                        // 4 MB
    __bf16* wqb  = vT + NE;                          // 0.5 MB (12.5 MB total ws)

    // Wq -> bf16 once
    cvt_wq_kernel<<<(HDIM * CDIM + 255) / 256, 256, 0, stream>>>(Wq, wqb);
    // q projection + RoPE (WMMA): 1024 waves = B*T/16 tiles
    qproj_rope_kernel<<<256, 128, 0, stream>>>(x, wqb, qrot);
    // RoPE on K -> bf16
    rope_k_kernel<<<(BATCH * TSEQ * (HDIM / 2) + 255) / 256, 256, 0, stream>>>(key, krot);
    // V -> V^T bf16
    transpose_v_kernel<<<(BATCH * HDIM * TSEQ + 255) / 256, 256, 0, stream>>>(val, vT);
    // flash attention (WMMA)
    attn_kernel<<<256, 128, 0, stream>>>(qrot, krot, vT, out);
}
